// aLSTM_35914516529898
// MI455X (gfx1250) — matrix operand order repaired and checked by reading, not compiled, not run on hardware
//
#include <hip/hip_runtime.h>
#include <cstdint>
#include <cstddef>

// ---------------------------------------------------------------------------
// Problem constants (from reference)
// ---------------------------------------------------------------------------
#define TT 512
#define BB 64
#define DD 1024
#define AA 128
#define KA 2304          // AIN (2176) + A (128): a_wih and a_whh folded on K
#define PSZ 14336        // 14*D
#define G4A 512          // 4*A
#define G4D 4096         // 4*D

#define WROW 40          // padded LDS row stride (elements) for 32-k chunks
#define PROW 136         // padded LDS row stride (elements) for pol panel

typedef __attribute__((ext_vector_type(16))) __bf16 v16bf;
typedef __attribute__((ext_vector_type(8)))  float  v8f;
typedef unsigned short u16;   // host-visible bf16 storage

// ---------------------------------------------------------------------------
// Fragment helpers
// ---------------------------------------------------------------------------
// 16-bit A-matrix 16x32 layout: lane m (=lane&15), khalf (=lane>>4):
//   elements 0..7  -> k = kbase + 8*khalf + j
//   elements 8..15 -> k = kbase + 16 + 8*khalf + j
union BFrag { uint4 u[2]; v16bf v; };

__device__ inline v16bf a_frag_f32(const float* __restrict__ p) {
    v16bf a;
#pragma unroll
    for (int j = 0; j < 8; ++j) {
        a[j]     = (__bf16)p[j];
        a[8 + j] = (__bf16)p[16 + j];
    }
    return a;
}

__device__ inline v16bf a_frag_f32_scaled(const float* __restrict__ p,
                                          const float* __restrict__ s) {
    v16bf a;
#pragma unroll
    for (int j = 0; j < 8; ++j) {
        a[j]     = (__bf16)(p[j] * s[j]);
        a[8 + j] = (__bf16)(p[16 + j] * s[16 + j]);
    }
    return a;
}

// direct bf16 A-frag: two 16B loads at +0 and +16 elements, no conversion
__device__ inline v16bf a_frag_bf16(const __bf16* __restrict__ p) {
    BFrag f;
    f.u[0] = *(const uint4*)(p);
    f.u[1] = *(const uint4*)(p + 16);
    return f.v;
}

// B-frag read from an LDS chunk: rows are (q*16+nt), padded stride WROW,
// lane reads 8 bf16 at k offset kh16/2 and 8 at kh16/2+16 -> two ds_load_b128.
__device__ inline v16bf b_frag_lds(const __bf16* __restrict__ s,
                                   int q, int nt, int kh16) {
    const __bf16* p = s + (q * 16 + nt) * WROW + (kh16 >> 1);
    BFrag f;
    f.u[0] = *(const uint4*)(p);
    f.u[1] = *(const uint4*)(p + 16);
    return f.v;
}

__device__ inline v8f wmma_bf16(v16bf a, v16bf b, v8f c) {
    return __builtin_amdgcn_wmma_f32_16x16x32_bf16(
        /*neg_a=*/false, a, /*neg_b=*/false, b,
        /*c_mod=*/(short)0, c, /*reuse_a=*/false, /*reuse_b=*/false);
}

__device__ inline float sigf(float x) { return 1.0f / (1.0f + __expf(-x)); }

// ---------------------------------------------------------------------------
// Cooperative weight staging: one 64-row x 32-k bf16 chunk (4KB) per step.
// Rows 0..63 map to (q = row>>4, r = row&15); global W row = q*qstride+nb+r.
// Thread tid (0..127): row = tid>>1, k-half = tid&1 (32B per thread).
// ---------------------------------------------------------------------------
__device__ inline void stage_w(const __bf16* __restrict__ W, int qstride, int ldk,
                               int nb, int kb, __bf16* __restrict__ s, int tid)
{
    const int row  = tid >> 1;          // 0..63
    const int half = tid & 1;           // k offset 0 / 16
    const int q = row >> 4, r = row & 15;
    const __bf16* g = W + (size_t)(q * qstride + nb + r) * ldk + kb + half * 16;
    uint4* d = (uint4*)(s + row * WROW + half * 16);
    d[0] = *(const uint4*)(g);
    d[1] = *(const uint4*)(g + 8);
}

// ---------------------------------------------------------------------------
// a-cell: g = [xc | fh | ah_other | ah_self] @ Wa.T + bih + bhh, LSTM gate
// Wa: 512 x 2304 bf16.  Grid: 8 blocks x 128 threads (4 waves).
// Weights staged through double-buffered LDS (shared by all 4 waves).
// XF32: xc segment is fp32 (layer 0 reads x); else bf16 shadow (layer 1).
// ---------------------------------------------------------------------------
template <bool XF32>
__global__ __launch_bounds__(128) void a_cell_kernel(
    const float* __restrict__ xc_f,    // B x D fp32 (XF32)
    const u16*   __restrict__ xc_b,    // B x D bf16 (!XF32)
    const u16*   __restrict__ fh_b,    // B x D bf16 shadow
    const u16*   __restrict__ ahx_b,   // B x A bf16 shadow (other layer)
    const u16*   __restrict__ ahs_b,   // B x A bf16 shadow (self, whh input)
    const float* __restrict__ ac_cur,  // B x A
    const u16*   __restrict__ Wa_,     // 512 x 2304 bf16
    const float* __restrict__ bih,     // 512
    const float* __restrict__ bhh,     // 512
    float* __restrict__ ah_nxt,        // B x A
    float* __restrict__ ac_nxt,        // B x A
    u16*   __restrict__ ahb_nxt)       // B x A bf16 shadow out
{
    __shared__ __bf16 sW[2][64 * WROW];

    const __bf16* Wa = (const __bf16*)Wa_;
    const int tid  = threadIdx.x;
    const int lane = tid & 31;
    const int wave = tid >> 5;
    const int nb   = blockIdx.x * 16;          // gate-col base within A
    const int nt   = lane & 15;
    const int kh8  = (lane >> 4) * 8;
    const int kh16 = (lane >> 4) * 16;
    const int row  = wave * 16 + nt;           // batch row for A-frag

    v8f acc[4] = {};

    stage_w(Wa, AA, KA, nb, 0, sW[0], tid);
    int step = 0;

    // segment 0: xc  (K offset 0)
    for (int k0 = 0; k0 < DD; k0 += 32, ++step) {
        __syncthreads();
        const int kb = step * 32;
        if (kb + 32 < KA) stage_w(Wa, AA, KA, nb, kb + 32, sW[(step + 1) & 1], tid);
        v16bf a = XF32
            ? a_frag_f32(xc_f + (size_t)row * DD + k0 + kh8)
            : a_frag_bf16((const __bf16*)xc_b + (size_t)row * DD + k0 + kh8);
        const __bf16* sb = sW[step & 1];
#pragma unroll
        for (int q = 0; q < 4; ++q)
            acc[q] = wmma_bf16(a, b_frag_lds(sb, q, nt, kh16), acc[q]);
    }
    // segment 1: fh  (K offset D)
    for (int k0 = 0; k0 < DD; k0 += 32, ++step) {
        __syncthreads();
        const int kb = step * 32;
        if (kb + 32 < KA) stage_w(Wa, AA, KA, nb, kb + 32, sW[(step + 1) & 1], tid);
        v16bf a = a_frag_bf16((const __bf16*)fh_b + (size_t)row * DD + k0 + kh8);
        const __bf16* sb = sW[step & 1];
#pragma unroll
        for (int q = 0; q < 4; ++q)
            acc[q] = wmma_bf16(a, b_frag_lds(sb, q, nt, kh16), acc[q]);
    }
    // segment 2: ah_other  (K offset 2D)
    for (int k0 = 0; k0 < AA; k0 += 32, ++step) {
        __syncthreads();
        const int kb = step * 32;
        if (kb + 32 < KA) stage_w(Wa, AA, KA, nb, kb + 32, sW[(step + 1) & 1], tid);
        v16bf a = a_frag_bf16((const __bf16*)ahx_b + (size_t)row * AA + k0 + kh8);
        const __bf16* sb = sW[step & 1];
#pragma unroll
        for (int q = 0; q < 4; ++q)
            acc[q] = wmma_bf16(a, b_frag_lds(sb, q, nt, kh16), acc[q]);
    }
    // segment 3: ah_self (whh)  (K offset 2176)
    for (int k0 = 0; k0 < AA; k0 += 32, ++step) {
        __syncthreads();
        const int kb = step * 32;
        if (kb + 32 < KA) stage_w(Wa, AA, KA, nb, kb + 32, sW[(step + 1) & 1], tid);
        v16bf a = a_frag_bf16((const __bf16*)ahs_b + (size_t)row * AA + k0 + kh8);
        const __bf16* sb = sW[step & 1];
#pragma unroll
        for (int q = 0; q < 4; ++q)
            acc[q] = wmma_bf16(a, b_frag_lds(sb, q, nt, kh16), acc[q]);
    }

    const int n = nb + nt;                 // [0,A)
    const float bi = bih[0 * AA + n] + bhh[0 * AA + n];
    const float bf = bih[1 * AA + n] + bhh[1 * AA + n];
    const float bg = bih[2 * AA + n] + bhh[2 * AA + n];
    const float bo = bih[3 * AA + n] + bhh[3 * AA + n];
#pragma unroll
    for (int v = 0; v < 8; ++v) {
        const int br = wave * 16 + 8 * (lane >> 4) + v;   // batch row (C layout)
        const float gi = acc[0][v] + bi;
        const float gf = acc[1][v] + bf;
        const float gg = acc[2][v] + bg;
        const float go = acc[3][v] + bo;
        const float c  = ac_cur[(size_t)br * AA + n];
        const float c2 = sigf(gf) * c + sigf(gi) * tanhf(gg);
        const float h2 = sigf(go) * tanhf(c2);
        ac_nxt[(size_t)br * AA + n]             = c2;
        ah_nxt[(size_t)br * AA + n]             = h2;
        ((__bf16*)ahb_nxt)[(size_t)br * AA + n] = (__bf16)h2;
    }
}

// ---------------------------------------------------------------------------
// pol = ah @ p_w.T + p_b.   Pw: 14336 x 128 bf16.
// Grid: 224 blocks x 128 threads; block owns 64 contiguous output cols.
// The block's whole 64x128 weight panel (16KB) is staged in LDS once.
// ---------------------------------------------------------------------------
__global__ __launch_bounds__(128) void pol_kernel(
    const u16*   __restrict__ ahb,   // B x A bf16 shadow
    const u16*   __restrict__ Pw_,   // PSZ x A bf16
    const float* __restrict__ pb,    // PSZ
    float* __restrict__ pol)         // B x PSZ
{
    __shared__ __bf16 sP[64 * PROW];

    const __bf16* Pw = (const __bf16*)Pw_;
    const int tid  = threadIdx.x;
    const int lane = tid & 31;
    const int wave = tid >> 5;
    const int nb   = blockIdx.x * 64;
    const int nt   = lane & 15;
    const int kh8  = (lane >> 4) * 8;
    const int kh16 = (lane >> 4) * 16;
    const int row  = wave * 16 + nt;

    // stage the 64x128 panel: thread -> (prow = tid>>1, 64-el half)
    {
        const int prow = tid >> 1, half = tid & 1;
        const __bf16* g = Pw + (size_t)(nb + prow) * AA + half * 64;
        uint4* d = (uint4*)(sP + prow * PROW + half * 64);
#pragma unroll
        for (int j = 0; j < 8; ++j) d[j] = *(const uint4*)(g + j * 8);
    }
    __syncthreads();

    v8f acc[4] = {};
#pragma unroll
    for (int kb = 0; kb < AA; kb += 32) {
        v16bf a = a_frag_bf16((const __bf16*)ahb + (size_t)row * AA + kb + kh8);
#pragma unroll
        for (int c = 0; c < 4; ++c) {
            const __bf16* p = sP + (c * 16 + nt) * PROW + kb + kh8;
            BFrag f;
            f.u[0] = *(const uint4*)(p);
            f.u[1] = *(const uint4*)(p + 16);
            acc[c] = wmma_bf16(a, f.v, acc[c]);
        }
    }
#pragma unroll
    for (int c = 0; c < 4; ++c) {
        const int n = nb + c * 16 + nt;
        const float bias = pb[n];
#pragma unroll
        for (int v = 0; v < 8; ++v) {
            const int br = wave * 16 + 8 * (lane >> 4) + v;
            pol[(size_t)br * PSZ + n] = acc[c][v] + bias;
        }
    }
}

// ---------------------------------------------------------------------------
// f-cell: ig = (xc*p0)@Wih.T * p2 + fb*p4 ; hg = (fh*p1)@Whh.T * p3
//         g = ig + hg ; LSTM gate -> fh/fc (+ bf16 shadow, optional ys out)
// Wih/Whh: 4096 x 1024 bf16. Grid: 64 blocks x 128 threads.
// Unified 64-step loop (steps 0..31 = Wih/xc*p0, 32..63 = Whh/fh*p1) with
// double-buffered LDS weight staging pipelined across the phase boundary.
// pol row layout: p0 [0,1024) p1 [1024,2048) p2 [2048,6144)
//                 p3 [6144,10240) p4 [10240,14336)
// ---------------------------------------------------------------------------
__global__ __launch_bounds__(128) void f_cell_kernel(
    const float* __restrict__ xc,      // B x D fp32
    const float* __restrict__ fh_cur,  // B x D fp32
    const float* __restrict__ fc_cur,  // B x D
    const float* __restrict__ pol,     // B x PSZ
    const u16*   __restrict__ Wih_,    // 4D x D bf16
    const u16*   __restrict__ Whh_,    // 4D x D bf16
    const float* __restrict__ fb,      // 4D
    float* __restrict__ fh_nxt,        // B x D
    float* __restrict__ fc_nxt,        // B x D
    u16*   __restrict__ fhb_nxt,       // B x D bf16 shadow out
    float* __restrict__ out)           // B x D slice of ys, or null
{
    __shared__ __bf16 sW[2][64 * WROW];

    const __bf16* Wih = (const __bf16*)Wih_;
    const __bf16* Whh = (const __bf16*)Whh_;
    const int tid  = threadIdx.x;
    const int lane = tid & 31;
    const int wave = tid >> 5;
    const int nb   = blockIdx.x * 16;
    const int nt   = lane & 15;
    const int kh8  = (lane >> 4) * 8;
    const int kh16 = (lane >> 4) * 16;
    const int row  = wave * 16 + nt;
    const float* polrow = pol + (size_t)row * PSZ;

    v8f accx[4] = {};
    v8f acch[4] = {};

    stage_w(Wih, DD, DD, nb, 0, sW[0], tid);

    for (int s = 0; s < 64; ++s) {
        __syncthreads();
        if (s + 1 < 64) {
            const __bf16* Wn = (s + 1 < 32) ? Wih : Whh;
            stage_w(Wn, DD, DD, nb, ((s + 1) & 31) * 32, sW[(s + 1) & 1], tid);
        }
        const int kb = (s & 31) * 32;
        const __bf16* sb = sW[s & 1];
        if (s < 32) {
            v16bf a = a_frag_f32_scaled(xc + (size_t)row * DD + kb + kh8,
                                        polrow + kb + kh8);
#pragma unroll
            for (int q = 0; q < 4; ++q)
                accx[q] = wmma_bf16(a, b_frag_lds(sb, q, nt, kh16), accx[q]);
        } else {
            v16bf a = a_frag_f32_scaled(fh_cur + (size_t)row * DD + kb + kh8,
                                        polrow + DD + kb + kh8);
#pragma unroll
            for (int q = 0; q < 4; ++q)
                acch[q] = wmma_bf16(a, b_frag_lds(sb, q, nt, kh16), acch[q]);
        }
    }

    const int n = nb + nt;  // [0,D)
    float fbq[4];
#pragma unroll
    for (int q = 0; q < 4; ++q) fbq[q] = fb[q * DD + n];

#pragma unroll
    for (int v = 0; v < 8; ++v) {
        const int br = wave * 16 + 8 * (lane >> 4) + v;
        const float* pr = pol + (size_t)br * PSZ;
        float g[4];
#pragma unroll
        for (int q = 0; q < 4; ++q) {
            const int gc = q * DD + n;
            g[q] = accx[q][v] * pr[2048 + gc]     // p2
                 + fbq[q]     * pr[10240 + gc]    // p4
                 + acch[q][v] * pr[6144 + gc];    // p3
        }
        const float c  = fc_cur[(size_t)br * DD + n];
        const float c2 = sigf(g[1]) * c + sigf(g[0]) * tanhf(g[2]);
        const float h2 = sigf(g[3]) * tanhf(c2);
        fc_nxt[(size_t)br * DD + n]             = c2;
        fh_nxt[(size_t)br * DD + n]             = h2;
        ((__bf16*)fhb_nxt)[(size_t)br * DD + n] = (__bf16)h2;
        if (out) out[(size_t)br * DD + n] = h2;
    }
}

// ---------------------------------------------------------------------------
// Weight conversion / init
// ---------------------------------------------------------------------------
__global__ void cvt_wa_kernel(const float* __restrict__ wih,   // 512 x 2176
                              const float* __restrict__ whh,   // 512 x 128
                              u16* __restrict__ dst)           // 512 x 2304 bf16
{
    int idx = blockIdx.x * blockDim.x + threadIdx.x;
    if (idx >= G4A * KA) return;
    int r = idx / KA, c = idx % KA;
    float v = (c < 2176) ? wih[(size_t)r * 2176 + c]
                         : whh[(size_t)r * AA + (c - 2176)];
    ((__bf16*)dst)[idx] = (__bf16)v;
}

__global__ void cvt_kernel(const float* __restrict__ src, u16* __restrict__ dst, int n) {
    int i = blockIdx.x * blockDim.x + threadIdx.x;
    if (i < n) ((__bf16*)dst)[i] = (__bf16)src[i];
}

__global__ void zero_kernel(float* __restrict__ p, int n) {
    int i = blockIdx.x * blockDim.x + threadIdx.x;
    if (i < n) p[i] = 0.0f;
}

__global__ void zero_u16_kernel(u16* __restrict__ p, int n) {
    int i = blockIdx.x * blockDim.x + threadIdx.x;
    if (i < n) p[i] = 0;           // bf16 +0.0
}

// ---------------------------------------------------------------------------
// Host orchestration
// ---------------------------------------------------------------------------
extern "C" void kernel_launch(void* const* d_in, const int* in_sizes, int n_in,
                              void* d_out, int out_size, void* d_ws, size_t ws_size,
                              hipStream_t stream)
{
    (void)in_sizes; (void)n_in; (void)out_size; (void)ws_size;

    const float* x = (const float*)d_in[0];
    const float* a_wih[2] = {(const float*)d_in[1],  (const float*)d_in[10]};
    const float* a_whh[2] = {(const float*)d_in[2],  (const float*)d_in[11]};
    const float* a_bih[2] = {(const float*)d_in[3],  (const float*)d_in[12]};
    const float* a_bhh[2] = {(const float*)d_in[4],  (const float*)d_in[13]};
    const float* p_w[2]   = {(const float*)d_in[5],  (const float*)d_in[14]};
    const float* p_b[2]   = {(const float*)d_in[6],  (const float*)d_in[15]};
    const float* f_wih[2] = {(const float*)d_in[7],  (const float*)d_in[16]};
    const float* f_whh[2] = {(const float*)d_in[8],  (const float*)d_in[17]};
    const float* f_b[2]   = {(const float*)d_in[9],  (const float*)d_in[18]};

    char* ws = (char*)d_ws;
    size_t off = 0;
    auto alloc = [&](size_t bytes) -> char* {
        char* p = ws + off;
        off = (off + bytes + 255) & ~(size_t)255;
        return p;
    };

    u16* Wa[2];  u16* Pw[2];  u16* Fwih[2];  u16* Fwhh[2];
    for (int l = 0; l < 2; ++l) {
        Wa[l]   = (u16*)alloc((size_t)G4A * KA * 2);
        Pw[l]   = (u16*)alloc((size_t)PSZ * AA * 2);
        Fwih[l] = (u16*)alloc((size_t)G4D * DD * 2);
        Fwhh[l] = (u16*)alloc((size_t)G4D * DD * 2);
    }
    float* pol = (float*)alloc((size_t)BB * PSZ * 4);

    // fp32 state block: [parity][layer]: ah, ac (B x A); fh, fc (B x D)
    const int ah_n = BB * AA, fh_n = BB * DD;
    const int state_n = 2 * 2 * (2 * ah_n + 2 * fh_n);
    float* states = (float*)alloc((size_t)state_n * 4);
    float* ah_[2][2]; float* ac_[2][2]; float* fh_[2][2]; float* fc_[2][2];
    {
        float* p = states;
        for (int par = 0; par < 2; ++par)
            for (int l = 0; l < 2; ++l) {
                ah_[par][l] = p; p += ah_n;
                ac_[par][l] = p; p += ah_n;
                fh_[par][l] = p; p += fh_n;
                fc_[par][l] = p; p += fh_n;
            }
    }

    // bf16 shadow block: [parity][layer]: ah_bf (B x A), fh_bf (B x D)
    const int shadow_n = 2 * 2 * (ah_n + fh_n);
    u16* shadows = (u16*)alloc((size_t)shadow_n * 2);
    u16* ahb_[2][2]; u16* fhb_[2][2];
    {
        u16* p = shadows;
        for (int par = 0; par < 2; ++par)
            for (int l = 0; l < 2; ++l) {
                ahb_[par][l] = p; p += ah_n;
                fhb_[par][l] = p; p += fh_n;
            }
    }

    // ---- one-time prep: bf16 weights + zeroed states -----------------------
    const int THR = 256;
    for (int l = 0; l < 2; ++l) {
        cvt_wa_kernel<<<(G4A * KA + THR - 1) / THR, THR, 0, stream>>>(
            a_wih[l], a_whh[l], Wa[l]);
        cvt_kernel<<<(PSZ * AA + THR - 1) / THR, THR, 0, stream>>>(
            p_w[l], Pw[l], PSZ * AA);
        cvt_kernel<<<(G4D * DD + THR - 1) / THR, THR, 0, stream>>>(
            f_wih[l], Fwih[l], G4D * DD);
        cvt_kernel<<<(G4D * DD + THR - 1) / THR, THR, 0, stream>>>(
            f_whh[l], Fwhh[l], G4D * DD);
    }
    zero_kernel<<<(state_n + THR - 1) / THR, THR, 0, stream>>>(states, state_n);
    zero_u16_kernel<<<(shadow_n + THR - 1) / THR, THR, 0, stream>>>(shadows, shadow_n);

    // ---- sequential scan over timesteps -----------------------------------
    for (int t = 0; t < TT; ++t) {
        const int cur = t & 1, nxt = cur ^ 1;
        const float* xc0 = x + (size_t)t * BB * DD;
        for (int l = 0; l < 2; ++l) {
            if (l == 0) {
                // xc = x[t] (fp32); ah_other = layer-1 ah from prev step
                a_cell_kernel<true><<<AA / 16, 128, 0, stream>>>(
                    xc0, nullptr,
                    fhb_[cur][0], ahb_[cur][1], ahb_[cur][0],
                    ac_[cur][0], Wa[0], a_bih[0], a_bhh[0],
                    ah_[nxt][0], ac_[nxt][0], ahb_[nxt][0]);
            } else {
                // xc = fh[0] updated this step (bf16 shadow);
                // ah_other = layer-0 ah updated this step
                a_cell_kernel<false><<<AA / 16, 128, 0, stream>>>(
                    nullptr, fhb_[nxt][0],
                    fhb_[cur][1], ahb_[nxt][0], ahb_[cur][1],
                    ac_[cur][1], Wa[1], a_bih[1], a_bhh[1],
                    ah_[nxt][1], ac_[nxt][1], ahb_[nxt][1]);
            }

            pol_kernel<<<PSZ / 64, 128, 0, stream>>>(
                ahb_[nxt][l], Pw[l], p_b[l], pol);

            const float* xcf = (l == 0) ? xc0 : fh_[nxt][0];
            float* outp = (l == 1) ? ((float*)d_out + (size_t)t * BB * DD) : nullptr;
            f_cell_kernel<<<DD / 16, 128, 0, stream>>>(
                xcf, fh_[cur][l], fc_[cur][l], pol,
                Fwih[l], Fwhh[l], f_b[l],
                fh_[nxt][l], fc_[nxt][l], fhb_[nxt][l], outp);
        }
    }
}
